// TranceptionBlockAttention_24824910971363
// MI455X (gfx1250) — matrix operand order repaired and checked by reading, not compiled, not run on hardware
//
#include <hip/hip_runtime.h>
#include <stdint.h>

// ---------------------------------------------------------------------------
// Tranception attention block for MI455X (gfx1250, wave32, WMMA).
// All matmuls use v_wmma_f32_16x16x32_bf16 (bf16 in, fp32 accumulate).
// Softmax row-max + exp in fp32; softmax row-sum via a ones-matrix WMMA so the
// denominator sums exactly the bf16 P that feeds the P*V numerator.
// ---------------------------------------------------------------------------

typedef __attribute__((ext_vector_type(16))) __bf16 v16bf;
typedef __attribute__((ext_vector_type(8)))  __bf16 v8bf;
typedef __attribute__((ext_vector_type(8)))  float  v8f;

#define Bb 2
#define Ss 2048
#define Ee 1024
#define Hh 16
#define Dd 64

__device__ inline unsigned short f2bf(float f) {
  union { float f; unsigned int u; } x; x.f = f;
  unsigned int u = x.u;
  u += 0x7fffu + ((u >> 16) & 1u);      // round-to-nearest-even
  return (unsigned short)(u >> 16);
}

__device__ inline v8f wmma_bf16(v16bf a, v16bf b, v8f c) {
  // D = A(16x32 bf16) * B(32x16 bf16) + C(16x16 f32)
  return __builtin_amdgcn_wmma_f32_16x16x32_bf16(
      false, a, false, b, (short)0, c, false, false);
}

__device__ inline v16bf cat16(v8bf lo, v8bf hi) {
  return __builtin_shufflevector(lo, hi, 0,1,2,3,4,5,6,7,8,9,10,11,12,13,14,15);
}

__device__ inline v16bf ones16() {
  v16bf o;
  #pragma unroll
  for (int i = 0; i < 16; ++i) o[i] = (__bf16)1.0f;
  return o;
}

__device__ inline float redmax16(float x) {
  x = fmaxf(x, __shfl_xor(x, 1, 32));
  x = fmaxf(x, __shfl_xor(x, 2, 32));
  x = fmaxf(x, __shfl_xor(x, 4, 32));
  x = fmaxf(x, __shfl_xor(x, 8, 32));
  return x;   // reduced within each 16-lane half (matches C/D layout halves)
}

// ---------------------------------------------------------------------------
// GEMM: C[M,N] = A[M,K] * B[K,N] + bias[N].  A fp32 or bf16 (A_BF16), B fp32.
// Block: 256 threads (8 waves), tile 128x64, K-panel 64 (8 WMMA per sync).
// ---------------------------------------------------------------------------
template<int A_BF16>
__global__ __launch_bounds__(256)
void gemm_wmma(const void* __restrict__ Ain, const float* __restrict__ Bin,
               const float* __restrict__ bias, float* __restrict__ Cout,
               int M, int N, int K)
{
  __shared__ unsigned short As[128 * 64];   // [row][k]  bf16  (16 KB)
  __shared__ unsigned short Bs[64 * 64];    // [n][k]    bf16  (8 KB, transposed)

  const int tid  = threadIdx.x;
  const int lane = tid & 31, w = tid >> 5;
  const int m    = lane & 15, hf = lane >> 4;
  const int Mb   = blockIdx.x * 128;
  const int Nb   = blockIdx.y * 64;

  v8f acc0 = {}, acc1 = {}, acc2 = {}, acc3 = {};

  for (int k0 = 0; k0 < K; k0 += 64) {
    __syncthreads();
    // ---- stage A panel (128x64) into LDS as bf16 ----
    #pragma unroll
    for (int it = 0; it < 8; ++it) {
      int row = (tid >> 4) + it * 16;
      int kk  = (tid & 15) * 4;
      ushort4 sv;
      if (A_BF16) {
        sv = *(const ushort4*)((const unsigned short*)Ain +
                               (size_t)(Mb + row) * K + k0 + kk);
      } else {
        float4 fv = *(const float4*)((const float*)Ain +
                                     (size_t)(Mb + row) * K + k0 + kk);
        sv.x = f2bf(fv.x); sv.y = f2bf(fv.y);
        sv.z = f2bf(fv.z); sv.w = f2bf(fv.w);
      }
      *(ushort4*)(&As[row * 64 + kk]) = sv;
    }
    // ---- stage B panel (64x64) transposed into LDS as bf16 ----
    #pragma unroll
    for (int it = 0; it < 4; ++it) {
      int kr = (tid >> 4) + it * 16;
      int nn = (tid & 15) * 4;
      float4 fv = *(const float4*)(Bin + (size_t)(k0 + kr) * N + Nb + nn);
      Bs[(nn + 0) * 64 + kr] = f2bf(fv.x);
      Bs[(nn + 1) * 64 + kr] = f2bf(fv.y);
      Bs[(nn + 2) * 64 + kr] = f2bf(fv.z);
      Bs[(nn + 3) * 64 + kr] = f2bf(fv.w);
    }
    if (k0 + 64 < K)   // pull next B panel toward L2 (global_prefetch_b8)
      __builtin_prefetch((const char*)Bin + ((size_t)(k0 + 64) * N + Nb) * 4, 0, 1);
    __syncthreads();

    // ---- 2 K-chunks x 4 col tiles = 8 WMMAs per panel ----
    #pragma unroll
    for (int c = 0; c < 2; ++c) {
      const unsigned short* ar = &As[(w * 16 + m) * 64 + c * 32];
      v16bf a = cat16(*(const v8bf*)(ar + hf * 8),
                      *(const v8bf*)(ar + 16 + hf * 8));
      const unsigned short* b0p = &Bs[( 0 + m) * 64 + c * 32 + hf * 8];
      const unsigned short* b1p = &Bs[(16 + m) * 64 + c * 32 + hf * 8];
      const unsigned short* b2p = &Bs[(32 + m) * 64 + c * 32 + hf * 8];
      const unsigned short* b3p = &Bs[(48 + m) * 64 + c * 32 + hf * 8];
      acc0 = wmma_bf16(a, cat16(*(const v8bf*)b0p, *(const v8bf*)(b0p + 16)), acc0);
      acc1 = wmma_bf16(a, cat16(*(const v8bf*)b1p, *(const v8bf*)(b1p + 16)), acc1);
      acc2 = wmma_bf16(a, cat16(*(const v8bf*)b2p, *(const v8bf*)(b2p + 16)), acc2);
      acc3 = wmma_bf16(a, cat16(*(const v8bf*)b3p, *(const v8bf*)(b3p + 16)), acc3);
    }
  }

  // ---- epilogue: +bias, fp32 store (C/D layout: VGPR r, half -> row) ----
  #pragma unroll
  for (int r = 0; r < 8; ++r) {
    int row = Mb + w * 16 + hf * 8 + r;
    float* cr = Cout + (size_t)row * N + Nb;
    cr[ 0 + m] = acc0[r] + bias[Nb +  0 + m];
    cr[16 + m] = acc1[r] + bias[Nb + 16 + m];
    cr[32 + m] = acc2[r] + bias[Nb + 32 + m];
    cr[48 + m] = acc3[r] + bias[Nb + 48 + m];
  }
}

// ---------------------------------------------------------------------------
// Depthwise causal conv per head-group + split to [B,H,S,D] bf16.
// t: 0=q (also folds 1/sqrt(D)), 1=k, 2=v.  Heads 0-3 pass through.
// ---------------------------------------------------------------------------
__global__ __launch_bounds__(256)
void conv_heads(const float* __restrict__ qkv,
  const float* qw3, const float* qb3, const float* qw5, const float* qb5,
  const float* qw7, const float* qb7,
  const float* kw3, const float* kb3, const float* kw5, const float* kb5,
  const float* kw7, const float* kb7,
  const float* vw3, const float* vb3, const float* vw5, const float* vb5,
  const float* vw7, const float* vb7,
  unsigned short* __restrict__ qo, unsigned short* __restrict__ ko,
  unsigned short* __restrict__ vo)
{
  int idx = blockIdx.x * 256 + threadIdx.x;
  int d = idx & 63;
  int s = (idx >> 6) & 2047;
  int h = (idx >> 17) & 15;
  int b = (idx >> 21) & 1;
  int t = idx >> 22;               // 0..2 (uniform per wave)
  int g = h >> 2;                  // head group

  const float* base = qkv + (size_t)(b * Ss) * 3 * Ee + t * Ee + h * Dd + d;
  float acc;
  if (g == 0) {
    acc = base[(size_t)s * 3 * Ee];
  } else {
    const float* wp; const float* bp;
    const int kern = 2 * g + 1;    // 3,5,7
    if (t == 0)      { wp = (g==1)?qw3:(g==2)?qw5:qw7; bp = (g==1)?qb3:(g==2)?qb5:qb7; }
    else if (t == 1) { wp = (g==1)?kw3:(g==2)?kw5:kw7; bp = (g==1)?kb3:(g==2)?kb5:kb7; }
    else             { wp = (g==1)?vw3:(g==2)?vw5:vw7; bp = (g==1)?vb3:(g==2)?vb5:vb7; }
    acc = bp[d];
    for (int i = 0; i < kern; ++i) {
      int ss = s - (kern - 1) + i;
      if (ss >= 0) acc += base[(size_t)ss * 3 * Ee] * wp[d * kern + i];
    }
  }
  if (t == 0) acc *= 0.125f;       // 1/sqrt(64) folded into q

  size_t o = ((size_t)((b * Hh + h) * Ss + s)) * Dd + d;
  if (t == 0)      qo[o] = f2bf(acc);
  else if (t == 1) ko[o] = f2bf(acc);
  else             vo[o] = f2bf(acc);
}

// ---------------------------------------------------------------------------
// Flash attention per (b,h): block = 128 query rows (8 waves x 16), key tiles
// of 32.  Causal mask = -10000 bias like the reference.  Output bf16 [B,S,E].
// ---------------------------------------------------------------------------
__global__ __launch_bounds__(256)
void attn_fwd(const unsigned short* __restrict__ q,
              const unsigned short* __restrict__ k,
              const unsigned short* __restrict__ v,
              unsigned short* __restrict__ o)
{
  __shared__ unsigned short Ks[32 * 64];     // [key][d]        bf16
  __shared__ unsigned short Vs[64 * 32];     // [d][key] (T)    bf16
  __shared__ unsigned short Pb[8][16 * 32];  // per-wave P bounce buffer

  const int tid  = threadIdx.x;
  const int lane = tid & 31, w = tid >> 5;
  const int m    = lane & 15, hf = lane >> 4;
  const int bh   = blockIdx.x;               // 0..31  (b*16+h)
  const int qb   = blockIdx.y;               // 0..15
  const int qbase = qb * 128 + w * 16;
  // wave-uniform scalar copy: keeps the causal skip on the SALU path so EXEC
  // stays all-1s at every WMMA (ISA requirement).
  const int qtop_s = __builtin_amdgcn_readfirstlane(qbase + 15);

  const unsigned short* qp = q + (size_t)bh * Ss * Dd;
  const unsigned short* kp = k + (size_t)bh * Ss * Dd;
  const unsigned short* vp = v + (size_t)bh * Ss * Dd;

  // Q fragments for K-chunks d=[0,32) and [32,64) (A 16x32 layout)
  v16bf aQ0, aQ1;
  {
    const unsigned short* pr = qp + (size_t)(qbase + m) * Dd + hf * 8;
    aQ0 = cat16(*(const v8bf*)(pr),      *(const v8bf*)(pr + 16));
    aQ1 = cat16(*(const v8bf*)(pr + 32), *(const v8bf*)(pr + 48));
  }
  const v16bf bOnes = ones16();

  v8f acc0 = {}, acc1 = {}, acc2 = {}, acc3 = {};
  float mrun[8], lrun[8];
  #pragma unroll
  for (int r = 0; r < 8; ++r) { mrun[r] = -1e30f; lrun[r] = 0.0f; }

  const int ntiles = (qb + 1) * 4;           // keys up to qb*128+127
  for (int jt = 0; jt < ntiles; ++jt) {
    const int kb = jt * 32;
    __syncthreads();
    { // cooperative stage of K (row-major) and V (transposed) tiles
      int e = tid * 8, key = e >> 6, d0 = e & 63;
      *(uint4*)(&Ks[key * 64 + d0]) =
          *(const uint4*)(kp + (size_t)(kb + key) * Dd + d0);
      union { uint4 u; unsigned short s[8]; } vv;
      vv.u = *(const uint4*)(vp + (size_t)(kb + key) * Dd + d0);
      #pragma unroll
      for (int j = 0; j < 8; ++j) Vs[(d0 + j) * 32 + key] = vv.s[j];
    }
    __syncthreads();

    if (kb <= qtop_s) {                      // scalar branch, EXEC untouched
      // ---- scores S = Q * K^T  (two 16x16 tiles, K-dim 64 = 2 chunks) ----
      v8f s0 = {}, s1 = {};
      #pragma unroll
      for (int c = 0; c < 2; ++c) {
        const unsigned short* b0p = &Ks[ m       * 64 + c * 32 + hf * 8];
        const unsigned short* b1p = &Ks[(16 + m) * 64 + c * 32 + hf * 8];
        v16bf b0 = cat16(*(const v8bf*)b0p, *(const v8bf*)(b0p + 16));
        v16bf b1 = cat16(*(const v8bf*)b1p, *(const v8bf*)(b1p + 16));
        v16bf aq = c ? aQ1 : aQ0;
        s0 = wmma_bf16(aq, b0, s0);
        s1 = wmma_bf16(aq, b1, s1);
      }
      // ---- causal mask + online max (fp32), write P (bf16) to LDS ----
      unsigned short* pb = &Pb[w][0];
      float sc[8];
      #pragma unroll
      for (int r = 0; r < 8; ++r) {
        const int row = qbase + hf * 8 + r;
        float x0 = (kb +      m <= row) ? s0[r] : -10000.0f;
        float x1 = (kb + 16 + m <= row) ? s1[r] : -10000.0f;
        float mx = redmax16(fmaxf(x0, x1));
        float mn = fmaxf(mrun[r], mx);
        sc[r] = __expf(mrun[r] - mn);
        mrun[r] = mn;
        pb[(hf * 8 + r) * 32 +      m] = f2bf(__expf(x0 - mn));
        pb[(hf * 8 + r) * 32 + 16 + m] = f2bf(__expf(x1 - mn));
        acc0[r] *= sc[r]; acc1[r] *= sc[r];
        acc2[r] *= sc[r]; acc3[r] *= sc[r];
      }
      // per-wave LDS bounce: D-layout -> A-layout; DS ops in order + wait
      asm volatile("s_wait_dscnt 0x0" ::: "memory");
      const unsigned short* pr = pb + m * 32 + hf * 8;
      v16bf aP = cat16(*(const v8bf*)pr, *(const v8bf*)(pr + 16));
      // ---- row-sum(P) via ones-matrix WMMA (replaces shuffle reduction) ----
      {
        v8f zero = {};
        v8f psum = wmma_bf16(aP, bOnes, zero);
        #pragma unroll
        for (int r = 0; r < 8; ++r) lrun[r] = lrun[r] * sc[r] + psum[r];
      }
      // ---- O += P * V  (4 output col tiles, K-dim 32) ----
      {
        const unsigned short* vb0 = &Vs[( 0 + m) * 32 + hf * 8];
        const unsigned short* vb1 = &Vs[(16 + m) * 32 + hf * 8];
        const unsigned short* vb2 = &Vs[(32 + m) * 32 + hf * 8];
        const unsigned short* vb3 = &Vs[(48 + m) * 32 + hf * 8];
        acc0 = wmma_bf16(aP, cat16(*(const v8bf*)vb0, *(const v8bf*)(vb0 + 16)), acc0);
        acc1 = wmma_bf16(aP, cat16(*(const v8bf*)vb1, *(const v8bf*)(vb1 + 16)), acc1);
        acc2 = wmma_bf16(aP, cat16(*(const v8bf*)vb2, *(const v8bf*)(vb2 + 16)), acc2);
        acc3 = wmma_bf16(aP, cat16(*(const v8bf*)vb3, *(const v8bf*)(vb3 + 16)), acc3);
      }
    }
  }

  // ---- normalize and store bf16 to [B,S,E] (head h -> cols h*64..h*64+63)
  const int b = bh >> 4, h = bh & 15;
  #pragma unroll
  for (int r = 0; r < 8; ++r) {
    const int row = qbase + hf * 8 + r;
    const float inv = 1.0f / lrun[r];
    unsigned short* orow = o + ((size_t)(b * Ss + row)) * Ee + h * Dd;
    orow[ 0 + m] = f2bf(acc0[r] * inv);
    orow[16 + m] = f2bf(acc1[r] * inv);
    orow[32 + m] = f2bf(acc2[r] * inv);
    orow[48 + m] = f2bf(acc3[r] * inv);
  }
}

// ---------------------------------------------------------------------------
extern "C" void kernel_launch(void* const* d_in, const int* in_sizes, int n_in,
                              void* d_out, int out_size, void* d_ws, size_t ws_size,
                              hipStream_t stream)
{
  (void)in_sizes; (void)n_in; (void)out_size; (void)ws_size;

  const float* hidden = (const float*)d_in[0];
  const float* attnw  = (const float*)d_in[1];
  const float* attnb  = (const float*)d_in[2];
  const float* projw  = (const float*)d_in[3];
  const float* projb  = (const float*)d_in[4];

  // Workspace layout (80 MB total):
  char* ws = (char*)d_ws;
  float*          qkv = (float*)(ws);                       // 4096*3072 f32 = 48 MB
  unsigned short* qh  = (unsigned short*)(ws + 50331648);   // [B,H,S,D] bf16, 8 MB
  unsigned short* kh  = (unsigned short*)(ws + 58720256);   // 8 MB
  unsigned short* vh  = (unsigned short*)(ws + 67108864);   // 8 MB
  unsigned short* ao  = (unsigned short*)(ws + 75497472);   // [B,S,E] bf16, 8 MB

  // 1) QKV projection: [4096,1024] x [1024,3072] + bias -> fp32 qkv
  gemm_wmma<0><<<dim3(32, 48), 256, 0, stream>>>(
      hidden, attnw, attnb, qkv, Bb * Ss, 3 * Ee, Ee);

  // 2) tranception depthwise conv + head split (q pre-scaled by 1/8), bf16 out
  conv_heads<<<(3 * Bb * Hh * Ss * Dd) / 256, 256, 0, stream>>>(
      qkv,
      (const float*)d_in[5],  (const float*)d_in[6],
      (const float*)d_in[7],  (const float*)d_in[8],
      (const float*)d_in[9],  (const float*)d_in[10],
      (const float*)d_in[11], (const float*)d_in[12],
      (const float*)d_in[13], (const float*)d_in[14],
      (const float*)d_in[15], (const float*)d_in[16],
      (const float*)d_in[17], (const float*)d_in[18],
      (const float*)d_in[19], (const float*)d_in[20],
      (const float*)d_in[21], (const float*)d_in[22],
      qh, kh, vh);

  // 3) causal flash attention per (b,h)
  attn_fwd<<<dim3(Bb * Hh, Ss / 128), 256, 0, stream>>>(qh, kh, vh, ao);

  // 4) output projection: [4096,1024] x [1024,1024] + bias -> fp32 d_out
  gemm_wmma<1><<<dim3(32, 16), 256, 0, stream>>>(
      ao, projw, projb, (float*)d_out, Bb * Ss, Ee, Ee);
}
